// RPQLinear_80917183856748
// MI455X (gfx1250) — compile-verified, matrix operand order checked by reading, not run on hardware
//
#include <hip/hip_runtime.h>

typedef __attribute__((ext_vector_type(16))) _Float16 v16h;
typedef __attribute__((ext_vector_type(8)))  _Float16 v8h;
typedef __attribute__((ext_vector_type(8)))  float    v8f;
typedef __attribute__((ext_vector_type(4)))  float    v4f;

// Problem dims (fixed by the reference)
#define M_TOT 8192     // B*S
#define N_TOT 16384    // V
#define K_TOT 4096     // H*D
#define H_CB  32
#define D_CB  128
#define V_CB  16384

// Tiling
#define BM  128
#define BN  128
#define KT  32         // K per stage (one f16 WMMA K-step)
#define LDA 40         // KT + 8 f16 pad -> 80B row stride, conflict-free b128 reads

__global__ __launch_bounds__(256)
void rpq_wmma_kernel(const float* __restrict__ x,
                     const int*   __restrict__ codes,
                     const float* __restrict__ codebooks,
                     const float* __restrict__ bias,
                     float*       __restrict__ out)
{
    // hi/lo split tiles: [precision][row][k]
    __shared__ __align__(16) _Float16 shA[2][BM][LDA];
    __shared__ __align__(16) _Float16 shB[2][BN][LDA];

    const int tid   = threadIdx.x;
    const int lane  = tid & 31;
    const int wave  = tid >> 5;       // 0..7
    const int waveM = wave >> 1;      // 0..3 -> 32 rows each
    const int waveN = wave & 1;       // 0..1 -> 64 cols each
    const int lrow  = lane & 15;      // row (A) / col (B) within 16x16 tile
    const int kgrp  = lane >> 4;      // K-group select per WMMA layout

    const int m0 = blockIdx.y * BM;
    const int n0 = blockIdx.x * BN;

    // staging mapping: 8 threads x float4 cover one 32-wide K slab per row
    const int c4    = (tid & 7) * 4;  // float offset in slab
    const int rbase = tid >> 3;       // 0..31, 4 passes of 32 rows

    // accumulators pre-loaded with bias (same bias for a whole C column)
    v8f acc[2][4];
    for (int tn = 0; tn < 4; ++tn) {
        const float bv = bias[n0 + waveN * 64 + tn * 16 + lrow];
        v8f c;
        for (int i = 0; i < 8; ++i) c[i] = bv;
        acc[0][tn] = c;
        acc[1][tn] = c;
    }

    // register stage for the software pipeline (global -> regs -> LDS)
    v4f regA[4];
    v4f regB[4];

    auto load_slab = [&](int k0) {
        #pragma unroll
        for (int pass = 0; pass < 4; ++pass) {
            const int r = rbase + pass * 32;
            regA[pass] = *(const v4f*)(x + (size_t)(m0 + r) * K_TOT + k0 + c4);
        }
        const int h  = k0 >> 7;           // codebook for this K slab
        const int d0 = k0 & (D_CB - 1);   // offset within the 128-wide row
        #pragma unroll
        for (int pass = 0; pass < 4; ++pass) {
            const int cidx = rbase + pass * 32;
            const int idx  = codes[h * V_CB + n0 + cidx];   // L2-resident
            regB[pass] = *(const v4f*)(codebooks +
                         ((size_t)(h * 256 + idx)) * D_CB + d0 + c4);
        }
    };

    load_slab(0);   // prime the pipeline

    for (int k0 = 0; k0 < K_TOT; k0 += KT) {
        __syncthreads();   // previous compute's LDS reads done

        // ---- convert staged registers fp32 -> f16 hi/lo and store to LDS ----
        #pragma unroll
        for (int pass = 0; pass < 4; ++pass) {
            const int r = rbase + pass * 32;
            #pragma unroll
            for (int j = 0; j < 4; ++j) {
                const float f = regA[pass][j];
                const _Float16 hi = (_Float16)f;
                const _Float16 lo = (_Float16)(f - (float)hi);
                shA[0][r][c4 + j] = hi;
                shA[1][r][c4 + j] = lo;
            }
        }
        #pragma unroll
        for (int pass = 0; pass < 4; ++pass) {
            const int cidx = rbase + pass * 32;
            #pragma unroll
            for (int j = 0; j < 4; ++j) {
                const float f = regB[pass][j];
                const _Float16 hi = (_Float16)f;
                const _Float16 lo = (_Float16)(f - (float)hi);
                shB[0][cidx][c4 + j] = hi;
                shB[1][cidx][c4 + j] = lo;
            }
        }
        __syncthreads();

        // ---- issue next slab's global loads; latency hides under WMMAs ----
        if (k0 + KT < K_TOT)
            load_slab(k0 + KT);

        // ---- load WMMA fragments from LDS (2x ds_load_b128 per fragment) ----
        // 16-bit A layout: lane(0-15)=row, dw0-3: K=kgrp*8..+7, dw4-7: K=16+kgrp*8..+7
        v16h afrag[2][2];  // [tm][hi/lo]
        v16h bfrag[4][2];  // [tn][hi/lo]
        #pragma unroll
        for (int tm = 0; tm < 2; ++tm)
            #pragma unroll
            for (int p = 0; p < 2; ++p) {
                const _Float16* base = &shA[p][waveM * 32 + tm * 16 + lrow][kgrp * 8];
                const v8h p0 = *(const v8h*)base;
                const v8h p1 = *(const v8h*)(base + 16);
                v16h t;
                for (int e = 0; e < 8; ++e) { t[e] = p0[e]; t[8 + e] = p1[e]; }
                afrag[tm][p] = t;
            }
        #pragma unroll
        for (int tn = 0; tn < 4; ++tn)
            #pragma unroll
            for (int p = 0; p < 2; ++p) {
                const _Float16* base = &shB[p][waveN * 64 + tn * 16 + lrow][kgrp * 8];
                const v8h p0 = *(const v8h*)base;
                const v8h p1 = *(const v8h*)(base + 16);
                v16h t;
                for (int e = 0; e < 8; ++e) { t[e] = p0[e]; t[8 + e] = p1[e]; }
                bfrag[tn][p] = t;
            }

        // ---- 24 WMMAs: hi*hi + hi*lo + lo*hi (~fp32 accuracy) ----
        #pragma unroll
        for (int tm = 0; tm < 2; ++tm)
            #pragma unroll
            for (int tn = 0; tn < 4; ++tn) {
                v8f c = acc[tm][tn];
                c = __builtin_amdgcn_wmma_f32_16x16x32_f16(
                        false, afrag[tm][0], false, bfrag[tn][0], (short)0, c, false, false);
                c = __builtin_amdgcn_wmma_f32_16x16x32_f16(
                        false, afrag[tm][0], false, bfrag[tn][1], (short)0, c, false, false);
                c = __builtin_amdgcn_wmma_f32_16x16x32_f16(
                        false, afrag[tm][1], false, bfrag[tn][0], (short)0, c, false, false);
                acc[tm][tn] = c;
            }
    }

    // ---- epilogue: C layout lane(0-15)=N, kgrp selects M 0-7 / 8-15 ----
    // NT stores: output is write-once; keep x/codebooks resident in the 192MB L2.
    #pragma unroll
    for (int tm = 0; tm < 2; ++tm)
        #pragma unroll
        for (int tn = 0; tn < 4; ++tn) {
            const int n  = n0 + waveN * 64 + tn * 16 + lrow;
            const int mb = m0 + waveM * 32 + tm * 16 + kgrp * 8;
            #pragma unroll
            for (int v = 0; v < 8; ++v)
                __builtin_nontemporal_store(acc[tm][tn][v],
                                            &out[(size_t)(mb + v) * N_TOT + n]);
        }
}

extern "C" void kernel_launch(void* const* d_in, const int* in_sizes, int n_in,
                              void* d_out, int out_size, void* d_ws, size_t ws_size,
                              hipStream_t stream) {
    (void)in_sizes; (void)n_in; (void)d_ws; (void)ws_size; (void)out_size;
    const float* x         = (const float*)d_in[0];
    const int*   codes     = (const int*)d_in[1];
    const float* codebooks = (const float*)d_in[2];
    const float* bias      = (const float*)d_in[3];
    float*       out       = (float*)d_out;

    dim3 grid(N_TOT / BN, M_TOT / BM);  // 128 x 64 blocks
    rpq_wmma_kernel<<<grid, 256, 0, stream>>>(x, codes, codebooks, bias, out);
}
